// SemanticAttention_13864154431686
// MI455X (gfx1250) — compile-verified, matrix operand order checked by reading, not run on hardware
//
#include <hip/hip_runtime.h>
#include <math.h>

#define BB 8
#define NN 4096
#define CC 384
#define KK 4

typedef __attribute__((ext_vector_type(16))) _Float16 v16h;
typedef __attribute__((ext_vector_type(8)))  _Float16 v8h;
typedef __attribute__((ext_vector_type(8)))  float    v8f;
typedef __attribute__((ext_vector_type(4)))  float    v4f;

__device__ __forceinline__ float dot4(v4f a, v4f b) {
  return a.x * b.x + a.y * b.y + a.z * b.z + a.w * b.w;
}

__device__ __forceinline__ v16h cat8(v8h lo, v8h hi) {
  return __builtin_shufflevector(lo, hi, 0, 1, 2, 3, 4, 5, 6, 7,
                                 8, 9, 10, 11, 12, 13, 14, 15);
}

// ---------------------------------------------------------------------------
// K0: convert wv, wres (C x C fp32, row-major [out][in]) to f16
// ---------------------------------------------------------------------------
__global__ void k0_convert(const float* __restrict__ wv,
                           const float* __restrict__ wres,
                           _Float16* __restrict__ wv_h,
                           _Float16* __restrict__ wres_h) {
  int i = blockIdx.x * blockDim.x + threadIdx.x;
  if (i < CC * CC) {
    wv_h[i]   = (_Float16)wv[i];
    wres_h[i] = (_Float16)wres[i];
  }
}

// ---------------------------------------------------------------------------
// K1: rank-4 Q/K projections (fp32 VALU; tiny: ~0.4 GFLOP total)
// ---------------------------------------------------------------------------
__global__ void k1_qk(const float* __restrict__ x,
                      const float* __restrict__ wq, const float* __restrict__ bq,
                      const float* __restrict__ wk, const float* __restrict__ bk,
                      float* __restrict__ segmap, float* __restrict__ kout) {
  int t = blockIdx.x * blockDim.x + threadIdx.x;   // token 0..B*N
  if (t >= BB * NN) return;
  const v4f* xr = (const v4f*)(x + (size_t)t * CC);
  const v4f* q4 = (const v4f*)wq;
  const v4f* k4 = (const v4f*)wk;
  float qa[KK] = {0.f, 0.f, 0.f, 0.f};
  float ka[KK] = {0.f, 0.f, 0.f, 0.f};
  for (int c4 = 0; c4 < CC / 4; ++c4) {
    v4f xv = xr[c4];
#pragma unroll
    for (int k = 0; k < KK; ++k) {
      qa[k] += dot4(xv, q4[k * (CC / 4) + c4]);
      ka[k] += dot4(xv, k4[k * (CC / 4) + c4]);
    }
  }
#pragma unroll
  for (int k = 0; k < KK; ++k) {
    segmap[(size_t)t * KK + k] = qa[k] + bq[k];
    kout[(size_t)t * KK + k]   = ka[k] + bk[k];
  }
}

// ---------------------------------------------------------------------------
// K2: V projection, WMMA f16: vt[b][c][n] = (x @ wv^T + bv), stored f16
//     TRANSPOSED so K3's B-operand loads are contiguous b128.
//     Block owns 64 tokens (4 M-tiles); wave owns 48 channels.
//     Single base pointers + compile-time offsets keep addresses in SGPR/imm.
// ---------------------------------------------------------------------------
__global__ void __launch_bounds__(256, 1)
k2_vproj(const float* __restrict__ x,
         const _Float16* __restrict__ wv_h,
         const float* __restrict__ bv,
         _Float16* __restrict__ vt) {
  const int b    = blockIdx.x / (NN / 64);
  const int n0   = (blockIdx.x % (NN / 64)) * 64;
  const int lane = threadIdx.x & 31;
  const int wid  = threadIdx.x >> 5;
  const int mrow = lane & 15;
  const int hi   = lane >> 4;
  const int lo_rel = hi ? 8 : 0;
  const int hi_rel = hi ? 24 : 16;
  const int chb  = wid * 48 + mrow;   // ch[t] = chb + t*16

  const float* xbase = x + ((size_t)b * NN + n0 + mrow) * CC;          // +mt*16*CC
  const v8h*   wb    = (const v8h*)(wv_h + (size_t)chb * CC);          // +t*16*CC/8
  _Float16*    vst   = vt + ((size_t)b * CC + chb) * NN + n0 + 8 * hi; // +t*16*NN+mt*16

  v8f acc[4][3] = {};

#pragma unroll 2
  for (int cb = 0; cb < CC; cb += 32) {
    v16h am[4];
#pragma unroll
    for (int mt = 0; mt < 4; ++mt) {
      const float* xr = xbase + (size_t)mt * (16 * CC);
      v4f x0 = *(const v4f*)(xr + cb + lo_rel);
      v4f x1 = *(const v4f*)(xr + cb + lo_rel + 4);
      v4f x2 = *(const v4f*)(xr + cb + hi_rel);
      v4f x3 = *(const v4f*)(xr + cb + hi_rel + 4);
#pragma unroll
      for (int i = 0; i < 4; ++i) {
        am[mt][i]      = (_Float16)x0[i];
        am[mt][4 + i]  = (_Float16)x1[i];
        am[mt][8 + i]  = (_Float16)x2[i];
        am[mt][12 + i] = (_Float16)x3[i];
      }
    }
#pragma unroll
    for (int t = 0; t < 3; ++t) {
      v16h bm = cat8(wb[(size_t)t * (16 * CC / 8) + (cb + lo_rel) / 8],
                     wb[(size_t)t * (16 * CC / 8) + (cb + hi_rel) / 8]);
#pragma unroll
      for (int mt = 0; mt < 4; ++mt)
        acc[mt][t] = __builtin_amdgcn_wmma_f32_16x16x32_f16(
            false, am[mt], false, bm, (short)0, acc[mt][t], false, false);
    }
  }
  // D layout: VGPR j -> token (j + 8*hi); column = mrow -> channel chb+t*16.
#pragma unroll
  for (int t = 0; t < 3; ++t) {
    float bias = bv[chb + t * 16];
#pragma unroll
    for (int mt = 0; mt < 4; ++mt) {
      v8h o;
#pragma unroll
      for (int j = 0; j < 8; ++j) o[j] = (_Float16)(acc[mt][t][j] + bias);
      *(v8h*)(vst + (size_t)t * (16 * NN) + mt * 16) = o;
    }
  }
}

// ---------------------------------------------------------------------------
// K3 (fused): flash attention + output projection + residual.
//   Block owns 64 queries (4 M-tiles of 16); 8 waves each own 48 channels.
//   Phase A: exact fp32 row max + sum(exp) over all 4096 keys (4 partials/q).
//   Phase B: 32 P tiles per 256-key super-iter built once (wave w builds
//            M-tile w>>1, k-steps (w&1)*4..+4) into LDS in per-lane A-matrix
//            layout; every V B-tile load feeds 4 WMMAs (4 M-tiles).
//   Phase C: 64x384 attention tile staged in LDS (padded rows, union'd with
//            the P-tile buffer), then out = gamma*(attn @ wres^T + bres) + x.
// ---------------------------------------------------------------------------
__global__ void __launch_bounds__(256, 1)
k3_attn(const float* __restrict__ q,       // seg_map (B,N,4)
        const float* __restrict__ kk,      // seg_ft  (B,N,4)
        const _Float16* __restrict__ vt,   // (B,C,N) f16
        const _Float16* __restrict__ wres_h,
        const float* __restrict__ bres,
        const float* __restrict__ gamma,
        const float* __restrict__ x,
        float* __restrict__ fout) {
  // pa: 32 P tiles (16x32 f16) in per-lane A layout        = 32 KB
  // as: 64 tokens x 384 ch, row stride 392 halfs           = 49 KB
  //     (196 dwords/row; 196 % 64 == 4 -> 16 concurrent rows hit
  //      16 distinct banks). Disjoint phases -> union.
  __shared__ union SH {
    v16h     pa[4][8][32];
    _Float16 as[64][392];
  } sh;
  __shared__ v4f   qsh[64];
  __shared__ float red[64][5];
  __shared__ float rmax64[64];
  __shared__ float rinv64[64];

  const int b   = blockIdx.x / (NN / 64);
  const int n0  = (blockIdx.x % (NN / 64)) * 64;
  const int tid = threadIdx.x;

  if (tid < 64) qsh[tid] = *(const v4f*)(q + ((size_t)b * NN + n0 + tid) * KK);
  __syncthreads();

  const v4f* kb4 = (const v4f*)(kk + (size_t)b * NN * KK);

  // ---- Phase A: per-query max, then sum of exp (4 partials per query) ----
  const int qi = tid >> 2, pp = tid & 3;
  {
    const v4f qv = qsh[qi];
    float lmax = -3.0e38f;
    for (int m = pp; m < NN; m += 4) lmax = fmaxf(lmax, dot4(qv, kb4[m]));
    red[qi][pp] = lmax;
    __syncthreads();
    if (pp == 0)
      rmax64[qi] = fmaxf(fmaxf(red[qi][0], red[qi][1]),
                         fmaxf(red[qi][2], red[qi][3]));
    __syncthreads();
    const float mx = rmax64[qi];
    float lsum = 0.f;
    for (int m = pp; m < NN; m += 4) lsum += __expf(dot4(qv, kb4[m]) - mx);
    red[qi][pp] = lsum;
    __syncthreads();
    if (pp == 0)
      rinv64[qi] = 1.0f / (red[qi][0] + red[qi][1] + red[qi][2] + red[qi][3]);
    __syncthreads();
  }

  // ---- Phase B: P @ V with WMMA ----
  const int lane = tid & 31, wid = tid >> 5;
  const int mrow = lane & 15, hi = lane >> 4;
  const int lo_rel = hi ? 8 : 0;
  const int hi_rel = hi ? 24 : 16;
  const int chb  = wid * 48 + mrow;   // ch[t] = chb + t*16

  // P-build assignment: this wave builds M-tile bmt, k-steps bio..bio+3
  const int  bmt  = wid >> 1;
  const int  bio  = (wid & 1) * 4;
  const int  bq   = bmt * 16 + mrow;
  const v4f  qm   = qsh[bq];
  const float mmax = rmax64[bq];
  const float minv = rinv64[bq];

  v8f acc[4][3] = {};
  const v8h* vb = (const v8h*)(vt + ((size_t)b * CC + chb) * NN); // +t*16*NN/8

  for (int kb0 = 0; kb0 < NN; kb0 += 256) {
    // build 4 of the 32 P tiles (exp computed exactly once per logit)
#pragma unroll
    for (int ii = 0; ii < 4; ++ii) {
      const v4f* kp = kb4 + kb0 + (bio + ii) * 32;
      v16h a;
#pragma unroll
      for (int i = 0; i < 8; ++i) {
        float l0 = dot4(qm, kp[lo_rel + i]);
        a[i]     = (_Float16)(__expf(l0 - mmax) * minv);
        float l1 = dot4(qm, kp[hi_rel + i]);
        a[8 + i] = (_Float16)(__expf(l1 - mmax) * minv);
      }
      sh.pa[bmt][bio + ii][lane] = a;
    }
    __syncthreads();

#pragma unroll 2
    for (int i = 0; i < 8; ++i) {
      const int kb = kb0 + i * 32;
      v16h bcur[3];
#pragma unroll
      for (int t = 0; t < 3; ++t)
        bcur[t] = cat8(vb[(size_t)t * (16 * NN / 8) + (kb + lo_rel) / 8],
                       vb[(size_t)t * (16 * NN / 8) + (kb + hi_rel) / 8]);
      v16h am[4];
#pragma unroll
      for (int mt = 0; mt < 4; ++mt) am[mt] = sh.pa[mt][i][lane];
#pragma unroll
      for (int t = 0; t < 3; ++t)
#pragma unroll
        for (int mt = 0; mt < 4; ++mt)
          acc[mt][t] = __builtin_amdgcn_wmma_f32_16x16x32_f16(
              false, am[mt], false, bcur[t], (short)0, acc[mt][t], false, false);
    }
    __syncthreads();
  }

  // ---- stage attention tile (1/rowsum already folded into P) ----
  {
    _Float16* asb = &sh.as[8 * hi][chb];   // +(mt*16+j)*392 + t*16
#pragma unroll
    for (int t = 0; t < 3; ++t)
#pragma unroll
      for (int mt = 0; mt < 4; ++mt)
#pragma unroll
        for (int j = 0; j < 8; ++j)
          asb[(mt * 16 + j) * 392 + t * 16] = (_Float16)acc[mt][t][j];
  }
  __syncthreads();

  // ---- Phase C: out = gamma * (attn @ wres^T + bres) + x ----
  v8f oacc[4][3] = {};
  const v8h* ar = (const v8h*)&sh.as[mrow][0];                 // +mt*16*49
  const v8h* wb = (const v8h*)(wres_h + (size_t)chb * CC);     // +t*16*CC/8

#pragma unroll 2
  for (int s = 0; s < CC / 32; ++s) {
    const int cb = s * 32;
    v16h bcur[3];
#pragma unroll
    for (int t = 0; t < 3; ++t)
      bcur[t] = cat8(wb[(size_t)t * (16 * CC / 8) + (cb + lo_rel) / 8],
                     wb[(size_t)t * (16 * CC / 8) + (cb + hi_rel) / 8]);
    v16h am[4];
#pragma unroll
    for (int mt = 0; mt < 4; ++mt)
      am[mt] = cat8(ar[(size_t)mt * (16 * 49) + (cb + lo_rel) / 8],
                    ar[(size_t)mt * (16 * 49) + (cb + hi_rel) / 8]);
#pragma unroll
    for (int t = 0; t < 3; ++t)
#pragma unroll
      for (int mt = 0; mt < 4; ++mt)
        oacc[mt][t] = __builtin_amdgcn_wmma_f32_16x16x32_f16(
            false, am[mt], false, bcur[t], (short)0, oacc[mt][t], false, false);
  }

  const float g = gamma[0];
  const size_t obase = ((size_t)b * NN + n0 + 8 * hi) * CC + chb;
#pragma unroll
  for (int t = 0; t < 3; ++t) {
    const float bias = bres[chb + t * 16];
#pragma unroll
    for (int mt = 0; mt < 4; ++mt)
#pragma unroll
      for (int j = 0; j < 8; ++j) {
        const size_t o = obase + (size_t)(mt * 16 + j) * CC + t * 16;
        fout[o] = g * (oacc[mt][t][j] + bias) + x[o];
      }
  }
}

// ---------------------------------------------------------------------------
extern "C" void kernel_launch(void* const* d_in, const int* in_sizes, int n_in,
                              void* d_out, int out_size, void* d_ws, size_t ws_size,
                              hipStream_t stream) {
  const float* x     = (const float*)d_in[0];
  const float* wq    = (const float*)d_in[1];
  const float* bq    = (const float*)d_in[2];
  const float* wk    = (const float*)d_in[3];
  const float* bk    = (const float*)d_in[4];
  const float* wv    = (const float*)d_in[5];
  const float* bv    = (const float*)d_in[6];
  const float* wres  = (const float*)d_in[7];
  const float* bres  = (const float*)d_in[8];
  const float* gamma = (const float*)d_in[9];

  float* out    = (float*)d_out;
  float* segmap = out;                           // (B,N,K)
  float* fout   = out + (size_t)BB * NN * KK;    // (B,N,C)

  // workspace layout (all chunks naturally 16B-aligned)
  _Float16* vt     = (_Float16*)d_ws;                          // B*C*N f16
  float*    k_ws   = (float*)(vt + (size_t)BB * CC * NN);      // B*N*K f32
  _Float16* wv_h   = (_Float16*)(k_ws + (size_t)BB * NN * KK); // C*C f16
  _Float16* wres_h = wv_h + (size_t)CC * CC;                   // C*C f16

  const int tiles64 = BB * (NN / 64);  // 512 blocks of 256 threads (8 waves)

  k0_convert<<<(CC * CC + 255) / 256, 256, 0, stream>>>(wv, wres, wv_h, wres_h);
  k1_qk<<<(BB * NN + 255) / 256, 256, 0, stream>>>(x, wq, bq, wk, bk, segmap, k_ws);
  k2_vproj<<<tiles64, 256, 0, stream>>>(x, wv_h, bv, vt);
  k3_attn<<<tiles64, 256, 0, stream>>>(segmap, k_ws, vt, wres_h, bres, gamma, x, fout);
}
